// ToMeResidualAttentionBlock_66666482369197
// MI455X (gfx1250) — matrix operand order repaired and checked by reading, not compiled, not run on hardware
//
#include <hip/hip_runtime.h>

// ---------------------------------------------------------------------------
// ToMe residual attention block for MI455X (gfx1250, wave32, WMMA).
// GEMMs: bf16 WMMA (v_wmma_f32_16x16x32_bf16), 32x32 tile per wave (2x2
// fragment reuse -> 4 independent WMMA chains, ~16 FLOP/byte from L2).
// Attention softmax + graph pooling stay in f32 VALU / LDS.
// ---------------------------------------------------------------------------

typedef unsigned short u16;
typedef __attribute__((ext_vector_type(16))) __bf16 v16bf;
typedef __attribute__((ext_vector_type(8)))  float  v8f;

static constexpr int Bb = 16;    // batch
static constexpr int Ss = 577;   // sequence (1 cls + 576 tokens)
static constexpr int Cc = 1024;  // channels
static constexpr int Hh = 16;    // heads
static constexpr int Dd = 64;    // head dim
static constexpr int Ee = 4608;  // edges
static constexpr int Nn = 576;   // tokens
static constexpr int Kt = 571;   // top-k kept tokens (ceil(0.99*576))
static constexpr int SP = 572;   // 1 + Kt
#define ATTN_SCALE 0.125f        // 64^-0.5

__device__ __forceinline__ u16 f2bf(float f) {
    unsigned u = __float_as_uint(f);
    u += 0x7FFFu + ((u >> 16) & 1u);       // round-to-nearest-even
    return (u16)(u >> 16);
}

union FragBF { v16bf v; uint4 q[2]; u16 h[16]; };

__device__ __forceinline__ v8f wmma_bf16(const FragBF& a, const FragBF& b, v8f c) {
    return __builtin_amdgcn_wmma_f32_16x16x32_bf16(
               false, a.v, false, b.v, (short)0, c, false, false);
}

// ---------------------------------------------------------------------------
// bf16 WMMA GEMM:  out[M,N] = A[M,K] @ W[N,K]^T  (+bias, +act, +resid)
// Block = 128 threads = 4 waves arranged 2x2; each wave owns a 32x32 tile
// (2 A-frags x 2 B-frags -> 4 WMMAs / K-step). N must be a multiple of 64;
// M rows are clamped/guarded. Fragment layouts per CDNA5 ISA 7.12.2.
// ---------------------------------------------------------------------------
template<int ACT, bool OUT_BF, bool HAS_RES>
__global__ __launch_bounds__(128)
void k_gemm(const u16* __restrict__ A, const u16* __restrict__ W,
            const float* __restrict__ bias, const float* __restrict__ res,
            float* __restrict__ outF, u16* __restrict__ outB,
            int M, int N, int K)
{
    const int wave = threadIdx.x >> 5;
    const int lane = threadIdx.x & 31;
    const int m0 = blockIdx.y * 64 + (wave >> 1) * 32;   // wave row tile
    const int n0 = blockIdx.x * 64 + (wave &  1) * 32;   // wave col tile

    const int l15 = lane & 15;
    const int ah  = (lane >> 4) * 8;    // A K-half: lanes 0-15 -> K+0..7/16..23
    const int bh  = (lane >> 4) * 8;    // B K-half: lanes 0-15 -> K+0..7/16..23

    int ra0 = m0 + l15;        if (ra0 > M - 1) ra0 = M - 1;   // clamp tail rows
    int ra1 = m0 + 16 + l15;   if (ra1 > M - 1) ra1 = M - 1;
    const u16* __restrict__ a0 = A + (size_t)ra0 * K;
    const u16* __restrict__ a1 = A + (size_t)ra1 * K;
    const u16* __restrict__ b0 = W + (size_t)(n0 + l15) * K;
    const u16* __restrict__ b1 = W + (size_t)(n0 + 16 + l15) * K;

    v8f acc00 = {}, acc01 = {}, acc10 = {}, acc11 = {};
    for (int k0 = 0; k0 < K; k0 += 32) {
        FragBF fa0, fa1, fb0, fb1;
        fa0.q[0] = *(const uint4*)(a0 + k0 + ah);
        fa0.q[1] = *(const uint4*)(a0 + k0 + 16 + ah);
        fa1.q[0] = *(const uint4*)(a1 + k0 + ah);
        fa1.q[1] = *(const uint4*)(a1 + k0 + 16 + ah);
        fb0.q[0] = *(const uint4*)(b0 + k0 + bh);
        fb0.q[1] = *(const uint4*)(b0 + k0 + 16 + bh);
        fb1.q[0] = *(const uint4*)(b1 + k0 + bh);
        fb1.q[1] = *(const uint4*)(b1 + k0 + 16 + bh);
        // prefetch next ~1KB of the streamed activation rows (weights are
        // L2-resident; A is touched once per GEMM) -> global_prefetch_b8
        __builtin_prefetch(a0 + k0 + 512, 0, 1);
        __builtin_prefetch(a1 + k0 + 512, 0, 1);
        // 4 independent accumulator chains hide WMMA->WMMA latency
        acc00 = wmma_bf16(fa0, fb0, acc00);
        acc01 = wmma_bf16(fa0, fb1, acc01);
        acc10 = wmma_bf16(fa1, fb0, acc10);
        acc11 = wmma_bf16(fa1, fb1, acc11);
    }

    const int nA = n0 + l15;            // C/D layout: lane = N, VGPR = M row
    const int mhalf = (lane >> 4) << 3; // lanes 16-31 hold M+8..15
    #pragma unroll
    for (int t = 0; t < 4; ++t) {
        const v8f& acc = (t == 0) ? acc00 : (t == 1) ? acc01
                        : (t == 2) ? acc10 : acc11;
        const int mt = m0 + ((t >> 1) << 4);
        const int n  = nA + ((t & 1) << 4);
        const int mb = mt + mhalf;
        const float bv = bias[n];
        #pragma unroll
        for (int r = 0; r < 8; ++r) {
            const int m = mb + r;
            if (m >= M) continue;
            float v = acc[r] + bv;
            if (ACT == 1) v = v * (1.f / (1.f + __expf(-1.702f * v)));
            if (HAS_RES) v += res[(size_t)m * N + n];
            const size_t o = (size_t)m * N + n;
            if (OUT_BF) outB[o] = f2bf(v);
            else        outF[o] = v;
        }
    }
}

// ---------------------------------------------------------------------------
// Elementwise f32 -> bf16 convert, and transpose-convert (for [K,N] weights).
// ---------------------------------------------------------------------------
__global__ void k_cvt(const float* __restrict__ s, u16* __restrict__ d, int n) {
    int i = blockIdx.x * 256 + threadIdx.x;
    if (i < n) d[i] = f2bf(s[i]);
}
__global__ void k_cvt_t(const float* __restrict__ s, u16* __restrict__ d,
                        int R, int Cn) {   // s[R,Cn] -> d[Cn,R]
    long long i = (long long)blockIdx.x * 256 + threadIdx.x;
    if (i < (long long)R * Cn) {
        int r = (int)(i / Cn), c = (int)(i % Cn);
        d[(size_t)c * R + r] = f2bf(s[i]);
    }
}

// ---------------------------------------------------------------------------
// LayerNorm over C=1024, one row per 256-thread block, bf16 output.
// ---------------------------------------------------------------------------
__global__ __launch_bounds__(256)
void k_ln(const float* __restrict__ x, const float* __restrict__ g,
          const float* __restrict__ b, u16* __restrict__ out)
{
    const int row = blockIdx.x;
    const float* xr = x + (size_t)row * Cc;
    __shared__ float sred[8];
    __shared__ float sm, sv;
    const int tid = threadIdx.x, lane = tid & 31, wv = tid >> 5;

    float s = 0.f;
    for (int c = tid; c < Cc; c += 256) s += xr[c];
    for (int o = 16; o; o >>= 1) s += __shfl_xor(s, o, 32);
    if (!lane) sred[wv] = s;
    __syncthreads();
    if (!tid) { float t = 0; for (int i = 0; i < 8; ++i) t += sred[i]; sm = t / Cc; }
    __syncthreads();
    const float mean = sm;

    s = 0.f;
    for (int c = tid; c < Cc; c += 256) { float d = xr[c] - mean; s += d * d; }
    for (int o = 16; o; o >>= 1) s += __shfl_xor(s, o, 32);
    if (!lane) sred[wv] = s;
    __syncthreads();
    if (!tid) { float t = 0; for (int i = 0; i < 8; ++i) t += sred[i]; sv = rsqrtf(t / Cc + 1e-5f); }
    __syncthreads();
    const float rinv = sv;

    for (int c = tid; c < Cc; c += 256)
        out[(size_t)row * Cc + c] = f2bf((xr[c] - mean) * rinv * g[c] + b[c]);
}

// ---------------------------------------------------------------------------
// Attention: one query row per wave32, logits staged in LDS, two-pass softmax
// with wave shuffles, key-bias = log(size). Writes bf16 context for out-proj.
// qkv layout: [B, S, 3C] with q|k|v at offsets 0|C|2C, head h at h*D.
// ---------------------------------------------------------------------------
__global__ __launch_bounds__(256)
void k_attn(const float* __restrict__ qkv, const float* __restrict__ sz,
            u16* __restrict__ obf)
{
    const int bh = blockIdx.x;            // B*H
    const int b = bh >> 4, h = bh & 15;
    const int wave = threadIdx.x >> 5, lane = threadIdx.x & 31;
    const int qr = blockIdx.y * 8 + wave;
    const bool valid = qr < Ss;

    __shared__ float sl[8 * Ss];
    __shared__ float sq[8][Dd];
    float* slw = sl + wave * Ss;
    const float* base = qkv + (size_t)b * Ss * (3 * Cc);

    if (valid) {
        const float* qp = base + (size_t)qr * (3 * Cc) + h * Dd;
        sq[wave][lane]      = qp[lane]      * ATTN_SCALE;
        sq[wave][lane + 32] = qp[lane + 32] * ATTN_SCALE;
    }
    __syncthreads();

    float mx = -1e30f;
    if (valid) {
        for (int j = lane; j < Ss; j += 32) {
            const float* kp = base + (size_t)j * (3 * Cc) + Cc + h * Dd;
            float a = 0.f;
            #pragma unroll 8
            for (int d = 0; d < Dd; ++d) a += sq[wave][d] * kp[d];
            a += __logf(sz[(size_t)b * Ss + j]);
            slw[j] = a;
            mx = fmaxf(mx, a);
        }
    }
    for (int o = 16; o; o >>= 1) mx = fmaxf(mx, __shfl_xor(mx, o, 32));
    __syncthreads();

    float sum = 0.f;
    if (valid) {
        for (int j = lane; j < Ss; j += 32) {
            float e = __expf(slw[j] - mx);
            slw[j] = e;
            sum += e;
        }
    }
    for (int o = 16; o; o >>= 1) sum += __shfl_xor(sum, o, 32);
    __syncthreads();

    if (valid) {
        const float inv = 1.f / sum;
        float a0 = 0.f, a1 = 0.f;
        for (int j = 0; j < Ss; ++j) {
            const float p = slw[j];
            const float* vp = base + (size_t)j * (3 * Cc) + 2 * Cc + h * Dd;
            a0 += p * vp[lane];
            a1 += p * vp[lane + 32];
        }
        u16* op = obf + ((size_t)b * Ss + qr) * Cc + h * Dd;
        op[lane]      = f2bf(a0 * inv);
        op[lane + 32] = f2bf(a1 * inv);
    }
}

// ---------------------------------------------------------------------------
// GCN score + exact top-K (rank matches jax.lax.top_k tie-breaking) + edge
// remap. One block of 576 threads per batch; deg/score via LDS atomics.
// ---------------------------------------------------------------------------
__global__ __launch_bounds__(576)
void k_pool(const float* __restrict__ x, const int* __restrict__ ei,
            const float* __restrict__ gw, const float* __restrict__ gb,
            int* __restrict__ map_ws, int* __restrict__ perm_ws,
            float* __restrict__ topv_ws,
            int* __restrict__ ei_out, float* __restrict__ mask_out)
{
    const int b = blockIdx.x, tid = threadIdx.x;
    __shared__ float sh[Nn], sdeg[Nn], ssc[Nn];
    __shared__ int   smap[Nn];
    const int* src = ei + (size_t)b * 2 * Ee;
    const int* dst = src + Ee;

    {   // h[n] = x_tok[n] . gcn_w + gcn_b
        const float* xr = x + ((size_t)b * Ss + 1 + tid) * Cc;
        float a = 0.f;
        for (int c = 0; c < Cc; ++c) a += xr[c] * gw[c];
        sh[tid]   = a + gb[0];
        sdeg[tid] = 1.f;                          // self-degree
    }
    __syncthreads();
    for (int e = tid; e < Ee; e += Nn) atomicAdd(&sdeg[dst[e]], 1.f);
    __syncthreads();
    sdeg[tid] = rsqrtf(sdeg[tid]);                // now dinv
    __syncthreads();
    ssc[tid] = sdeg[tid] * sdeg[tid] * sh[tid];   // self term
    __syncthreads();
    for (int e = tid; e < Ee; e += Nn) {
        const int s = src[e], d2 = dst[e];
        atomicAdd(&ssc[d2], sdeg[s] * sdeg[d2] * sh[s]);
    }
    __syncthreads();

    {   // rank[n] = #{m : score[m] > score[n]  or (==, m < n)}
        const float my = ssc[tid];
        int rank = 0;
        for (int m = 0; m < Nn; ++m) {
            const float o = ssc[m];
            rank += (o > my) || (o == my && m < tid);
        }
        const int mp = (rank < Kt) ? rank : -1;
        smap[tid] = mp;
        map_ws[(size_t)b * Nn + tid] = mp;
        if (mp >= 0) {
            perm_ws[(size_t)b * Kt + mp] = tid;
            topv_ws[(size_t)b * Kt + mp] = my;
        }
    }
    __syncthreads();

    for (int e = tid; e < Ee; e += Nn) {
        const int ns = smap[src[e]], nd = smap[dst[e]];
        const bool mk = (ns >= 0) && (nd >= 0);
        ei_out[((size_t)b * 2 + 0) * Ee + e] = mk ? ns : 0;
        ei_out[((size_t)b * 2 + 1) * Ee + e] = mk ? nd : 0;
        mask_out[(size_t)b * Ee + e] = mk ? 1.f : 0.f;
    }
}

// x2[b,0] = cls;  x2[b,1+j] = x[b,1+perm[j]] * tanh(topv[j])
__global__ __launch_bounds__(256)
void k_gather(const float* __restrict__ x, const int* __restrict__ perm,
              const float* __restrict__ topv, float* __restrict__ x2)
{
    const int b = blockIdx.y, j = blockIdx.x;
    const float* srcr;
    float scale = 1.f;
    if (j == 0) {
        srcr = x + (size_t)b * Ss * Cc;
    } else {
        const int p = perm[(size_t)b * Kt + (j - 1)];
        srcr  = x + ((size_t)b * Ss + 1 + p) * Cc;
        scale = tanhf(topv[(size_t)b * Kt + (j - 1)]);
    }
    float* dr = x2 + ((size_t)b * SP + j) * Cc;
    for (int c = threadIdx.x; c < Cc; c += 256) dr[c] = srcr[c] * scale;
}

// ---------------------------------------------------------------------------
extern "C" void kernel_launch(void* const* d_in, const int* in_sizes, int n_in,
                              void* d_out, int out_size, void* d_ws, size_t ws_size,
                              hipStream_t stream)
{
    const float* q_x   = (const float*)d_in[0];
    const int*   eidx  = (const int*)  d_in[1];
    const float* szs   = (const float*)d_in[2];
    const float* ln1_g = (const float*)d_in[3];
    const float* ln1_b = (const float*)d_in[4];
    const float* qkv_w = (const float*)d_in[5];
    const float* qkv_b = (const float*)d_in[6];
    const float* out_w = (const float*)d_in[7];
    const float* out_b = (const float*)d_in[8];
    const float* ln2_g = (const float*)d_in[9];
    const float* ln2_b = (const float*)d_in[10];
    const float* w1    = (const float*)d_in[11];
    const float* b1    = (const float*)d_in[12];
    const float* w2    = (const float*)d_in[13];
    const float* b2    = (const float*)d_in[14];
    const float* gw    = (const float*)d_in[15];
    const float* gb    = (const float*)d_in[16];

    char*  ws  = (char*)d_ws;
    size_t off = 0;
    auto alloc = [&](size_t bytes) -> char* {
        char* p = ws + off;
        off += (bytes + 255) & ~(size_t)255;
        return p;
    };
    const size_t MS = (size_t)Bb * Ss;   // 9232 rows (attention path)
    const size_t MP = (size_t)Bb * SP;   // 9152 rows (MLP path)

    u16*   qkvw_bf = (u16*)  alloc((size_t)3 * Cc * Cc * 2);
    u16*   outw_bf = (u16*)  alloc((size_t)Cc * Cc * 2);
    u16*   w1t_bf  = (u16*)  alloc((size_t)Cc * 4 * Cc * 2);   // [4C, C]
    u16*   w2t_bf  = (u16*)  alloc((size_t)4 * Cc * Cc * 2);   // [C, 4C]
    u16*   xn_bf   = (u16*)  alloc(MS * Cc * 2);
    float* qkv_f   = (float*)alloc(MS * 3 * Cc * 4);
    u16*   o_bf    = (u16*)  alloc(MS * Cc * 2);
    float* x_f     = (float*)alloc(MS * Cc * 4);
    float* x2_f    = (float*)alloc(MP * Cc * 4);
    u16*   xn2_bf  = (u16*)  alloc(MP * Cc * 2);
    u16*   hdn_bf  = (u16*)  alloc(MP * 4 * Cc * 2);
    int*   map_ws  = (int*)  alloc((size_t)Bb * Nn * 4);
    int*   perm_ws = (int*)  alloc((size_t)Bb * Kt * 4);
    float* topv_ws = (float*)alloc((size_t)Bb * Kt * 4);

    float* xo       = (float*)d_out;                                 // [B,SP,C]
    int*   ei_out   = (int*)((float*)d_out + MP * Cc);               // [B,2,E]
    float* mask_out = (float*)d_out + MP * Cc + (size_t)Bb * 2 * Ee; // [B,E]

    // Weight conversion (bf16; transpose [K,N] weights to [N,K] for WMMA-B).
    { int n = 3 * Cc * Cc; k_cvt<<<(n + 255) / 256, 256, 0, stream>>>(qkv_w, qkvw_bf, n); }
    { int n = Cc * Cc;     k_cvt<<<(n + 255) / 256, 256, 0, stream>>>(out_w, outw_bf, n); }
    { long long t = (long long)Cc * 4 * Cc;
      k_cvt_t<<<(unsigned)((t + 255) / 256), 256, 0, stream>>>(w1, w1t_bf, Cc, 4 * Cc);
      k_cvt_t<<<(unsigned)((t + 255) / 256), 256, 0, stream>>>(w2, w2t_bf, 4 * Cc, Cc); }

    // LN1 -> bf16
    k_ln<<<(unsigned)MS, 256, 0, stream>>>(q_x, ln1_g, ln1_b, xn_bf);

    // qkv = xn @ qkv_w^T + qkv_b      (M=9232, N=3072, K=1024)
    k_gemm<0, false, false><<<dim3(3 * Cc / 64, (MS + 63) / 64), 128, 0, stream>>>(
        xn_bf, qkvw_bf, qkv_b, nullptr, qkv_f, nullptr, (int)MS, 3 * Cc, Cc);

    // attention -> bf16 context
    k_attn<<<dim3(Bb * Hh, (Ss + 7) / 8), 256, 0, stream>>>(qkv_f, szs, o_bf);

    // x = q_x + o @ out_w^T + out_b   (M=9232, N=1024, K=1024)
    k_gemm<0, false, true><<<dim3(Cc / 64, (MS + 63) / 64), 128, 0, stream>>>(
        o_bf, outw_bf, out_b, q_x, x_f, nullptr, (int)MS, Cc, Cc);

    // graph pooling + edge outputs
    k_pool<<<Bb, Nn, 0, stream>>>(x_f, eidx, gw, gb, map_ws, perm_ws, topv_ws,
                                  ei_out, mask_out);
    k_gather<<<dim3(SP, Bb), 256, 0, stream>>>(x_f, perm_ws, topv_ws, x2_f);

    // LN2 -> bf16
    k_ln<<<(unsigned)MP, 256, 0, stream>>>(x2_f, ln2_g, ln2_b, xn2_bf);

    // hdn = act(xn2 @ mlp_w1 + b1)    (M=9152, N=4096, K=1024) -> bf16
    k_gemm<1, true, false><<<dim3(4 * Cc / 64, MP / 64), 128, 0, stream>>>(
        xn2_bf, w1t_bf, b1, nullptr, nullptr, hdn_bf, (int)MP, 4 * Cc, Cc);

    // out = x2 + hdn @ mlp_w2 + b2    (M=9152, N=1024, K=4096) -> d_out
    k_gemm<0, false, true><<<dim3(Cc / 64, MP / 64), 128, 0, stream>>>(
        hdn_bf, w2t_bf, b2, x2_f, xo, nullptr, (int)MP, Cc, 4 * Cc);
}
